// CustomMultiHeadAttention_5738076307699
// MI455X (gfx1250) — compile-verified, matrix operand order checked by reading, not run on hardware
//
#include <hip/hip_runtime.h>
#include <hip/hip_bf16.h>

typedef __attribute__((ext_vector_type(16))) __bf16 bf16x16;
typedef __attribute__((ext_vector_type(8)))  float  f32x8;
typedef int v4i_ __attribute__((vector_size(16)));   // matches builtin's pointee type

#define D_MODEL 768
#define NHEAD   12
#define DK      64
#define SEQ     4096
#define BATCH   2
#define MTOT    (BATCH*SEQ)   // 8192

#define GS 40   // GEMM LDS row stride (elems): 80B rows, 16B-aligned, bank stride 20
#define FS 72   // flash LDS row stride (elems): 144B rows, 16B-aligned, bank stride 36

#if defined(__gfx1250__) && __has_builtin(__builtin_amdgcn_global_load_async_to_lds_b128)
#define HAVE_ASYNC 1
#else
#define HAVE_ASYNC 0
#endif

// A/B fragment K-index mapping for v_wmma_*_16x16x32_bf16 (ISA 7.12.2):
// lanes 0-15 hold K 0..7 (elems 0..7) and 16..23 (elems 8..15);
// lanes 16-31 hold K 8..15 and 24..31.
__device__ __forceinline__ int kmap(int e, int kh) {
    return (e < 8) ? (kh * 8 + e) : (16 + kh * 8 + (e - 8));
}

__device__ __forceinline__ f32x8 wmma_bf16(const bf16x16& a, const bf16x16& b, const f32x8& c) {
    return __builtin_amdgcn_wmma_f32_16x16x32_bf16(false, a, false, b, (short)0, c, false, false);
}

__device__ __forceinline__ unsigned pack2bf(float a, float b) {
    union { __bf16 h[2]; unsigned u; } x;
    x.h[0] = (__bf16)a; x.h[1] = (__bf16)b;
    return x.u;
}

// 16-byte async global->LDS copy (ASYNCcnt path); no-op stub when unavailable.
__device__ __forceinline__ void async_copy16(const __bf16* g, __bf16* l) {
#if HAVE_ASYNC
    __builtin_amdgcn_global_load_async_to_lds_b128(
        (__attribute__((address_space(1))) v4i_*)g,
        (__attribute__((address_space(3))) v4i_*)l,
        0, 0);
#else
    (void)g; (void)l;
#endif
}

__device__ __forceinline__ void wait_async0() {
#if HAVE_ASYNC
#if __has_builtin(__builtin_amdgcn_s_wait_asynccnt)
    __builtin_amdgcn_s_wait_asynccnt(0);
#else
    asm volatile("s_wait_asynccnt 0x0" ::: "memory");
#endif
#endif
}

// ---------------------------------------------------------------------------
// Kernel 1: QKV projection. Y = X @ W.T + b  (W stored [out,in] row-major)
// Block: 256 thr = 8 waves. Tile: 128(M) x 64(N). grid = (64, 12 heads, 3 proj)
// Output layout: [b, h, s, dk] in bf16. (f32->bf16 conversion during staging,
// so this stays on the synchronous vectorized path.)
// ---------------------------------------------------------------------------
__global__ __launch_bounds__(256)
void qkv_proj_kernel(const float* __restrict__ X,
                     const float* __restrict__ Wq, const float* __restrict__ bq,
                     const float* __restrict__ Wk, const float* __restrict__ bk,
                     const float* __restrict__ Wv, const float* __restrict__ bv,
                     __bf16* __restrict__ Qb, __bf16* __restrict__ Kb, __bf16* __restrict__ Vb)
{
    const int proj = blockIdx.z;
    const float* W    = (proj == 0) ? Wq : (proj == 1) ? Wk : Wv;
    const float* bias = (proj == 0) ? bq : (proj == 1) ? bk : bv;
    __bf16*      Out  = (proj == 0) ? Qb : (proj == 1) ? Kb : Vb;

    const int m0 = blockIdx.x * 128;
    const int h  = blockIdx.y;          // 64-wide N tile == exactly one head
    const int n0 = h * DK;

    __shared__ __bf16 Xs[128 * GS];
    __shared__ __bf16 Ws[64 * GS];

    const int tid  = threadIdx.x;
    const int wave = tid >> 5;
    const int lane = tid & 31;
    const int lm   = lane & 15;
    const int kh   = lane >> 4;

    f32x8 acc[4] = {};

    for (int k0 = 0; k0 < D_MODEL; k0 += 32) {
        __syncthreads();
#pragma unroll
        for (int i = 0; i < 4; i++) {           // X tile 128x32 f32 -> packed bf16
            int lin = tid + i * 256;
            int r = lin >> 3, c = (lin & 7) * 4;
            const float4 v = *(const float4*)(X + (size_t)(m0 + r) * D_MODEL + k0 + c);
            uint2 p; p.x = pack2bf(v.x, v.y); p.y = pack2bf(v.z, v.w);
            *(uint2*)(&Xs[r * GS + c]) = p;
        }
#pragma unroll
        for (int i = 0; i < 2; i++) {           // W tile 64(n) x 32(k)
            int lin = tid + i * 256;
            int r = lin >> 3, c = (lin & 7) * 4;
            const float4 v = *(const float4*)(W + (size_t)(n0 + r) * D_MODEL + k0 + c);
            uint2 p; p.x = pack2bf(v.x, v.y); p.y = pack2bf(v.z, v.w);
            *(uint2*)(&Ws[r * GS + c]) = p;
        }
        __syncthreads();

        bf16x16 a;
        const int arow = wave * 16 + lm;
#pragma unroll
        for (int e = 0; e < 16; e++) a[e] = Xs[arow * GS + kmap(e, kh)];
#pragma unroll
        for (int t = 0; t < 4; t++) {
            bf16x16 b;
            const int bcol = t * 16 + lm;
#pragma unroll
            for (int e = 0; e < 16; e++) b[e] = Ws[bcol * GS + kmap(e, kh)];
            acc[t] = wmma_bf16(a, b, acc[t]);
        }
    }

#pragma unroll
    for (int t = 0; t < 4; t++) {
        const int d  = t * 16 + lm;
        const float bv_ = bias[n0 + d];
#pragma unroll
        for (int r = 0; r < 8; r++) {
            const int M    = m0 + wave * 16 + r + 8 * kh;
            const int bidx = M >> 12;           // / 4096
            const int s    = M & 4095;
            Out[(((size_t)bidx * NHEAD + h) * SEQ + s) * DK + d] = (__bf16)(acc[t][r] + bv_);
        }
    }
}

// ---------------------------------------------------------------------------
// Kernel 2: flash attention. Block: 128 queries x one (b,h); 8 waves x 16 rows.
// Online softmax in fp32; QK^T and P@V via bf16 WMMA. K/V tiles streamed with
// double-buffered async global->LDS copies (ASYNCcnt). grid = (32, 24)
// Context written bf16 in [B, S, D_MODEL] layout.
// ---------------------------------------------------------------------------
__global__ __launch_bounds__(256)
void flash_attn_kernel(const __bf16* __restrict__ Qb, const __bf16* __restrict__ Kb,
                       const __bf16* __restrict__ Vb, __bf16* __restrict__ Ctx)
{
    const int bh = blockIdx.y;
    const int q0 = blockIdx.x * 128;
    const __bf16* Qh = Qb + (size_t)bh * SEQ * DK;
    const __bf16* Kh = Kb + (size_t)bh * SEQ * DK;
    const __bf16* Vh = Vb + (size_t)bh * SEQ * DK;

    __shared__ __bf16 Qs[128 * FS];
    __shared__ __bf16 Ks[2][64 * FS];
    __shared__ __bf16 Vs[2][64 * FS];
    __shared__ __bf16 Ps[8 * 16 * FS];   // per-wave 16x64 P strip

    const int tid  = threadIdx.x;
    const int wave = tid >> 5;
    const int lane = tid & 31;
    const int lm   = lane & 15;
    const int kh   = lane >> 4;

    // this thread's two 16B chunks of each 64x64 tile
    const int cr0 = (tid)       >> 3, cc0 = ((tid)       & 7) * 8;
    const int cr1 = (tid + 256) >> 3, cc1 = ((tid + 256) & 7) * 8;

#if HAVE_ASYNC
    // prologue: start DMA of first K/V tiles while we stage Q
    async_copy16(Kh + (size_t)cr0 * DK + cc0, &Ks[0][cr0 * FS + cc0]);
    async_copy16(Vh + (size_t)cr0 * DK + cc0, &Vs[0][cr0 * FS + cc0]);
    async_copy16(Kh + (size_t)cr1 * DK + cc1, &Ks[0][cr1 * FS + cc1]);
    async_copy16(Vh + (size_t)cr1 * DK + cc1, &Vs[0][cr1 * FS + cc1]);
#endif

    // Q block 128x64 bf16: 1024 uint4 loads, 4 per thread, b128 LDS stores
#pragma unroll
    for (int i = 0; i < 4; i++) {
        int lin = tid + i * 256;
        int r = lin >> 3, c = (lin & 7) * 8;
        *(uint4*)(&Qs[r * FS + c]) = *(const uint4*)(Qh + (size_t)(q0 + r) * DK + c);
    }
    __syncthreads();

    bf16x16 qa[2];
    {
        const int arow = wave * 16 + lm;
#pragma unroll
        for (int dc = 0; dc < 2; dc++)
#pragma unroll
            for (int e = 0; e < 16; e++)
                qa[dc][e] = Qs[arow * FS + dc * 32 + kmap(e, kh)];
    }

    float mrow[8], lrow[8];
#pragma unroll
    for (int r = 0; r < 8; r++) { mrow[r] = -3.0e38f; lrow[r] = 0.0f; }
    f32x8 cacc[4] = {};

    __bf16* Pw = Ps + wave * 16 * FS;
    const float scale = 0.125f;                   // 1/sqrt(64)

    int buf = 0;
    for (int j0 = 0; j0 < SEQ; j0 += 64, buf ^= 1) {
#if HAVE_ASYNC
        wait_async0();          // my async copies into Ks/Vs[buf] done
        __syncthreads();        // everyone's done; prior readers of buf^1 done
        if (j0 + 64 < SEQ) {    // kick off next tile into the other buffer
            const int jn = j0 + 64;
            async_copy16(Kh + (size_t)(jn + cr0) * DK + cc0, &Ks[buf ^ 1][cr0 * FS + cc0]);
            async_copy16(Vh + (size_t)(jn + cr0) * DK + cc0, &Vs[buf ^ 1][cr0 * FS + cc0]);
            async_copy16(Kh + (size_t)(jn + cr1) * DK + cc1, &Ks[buf ^ 1][cr1 * FS + cc1]);
            async_copy16(Vh + (size_t)(jn + cr1) * DK + cc1, &Vs[buf ^ 1][cr1 * FS + cc1]);
        }
#else
        __syncthreads();
        *(uint4*)(&Ks[buf][cr0 * FS + cc0]) = *(const uint4*)(Kh + (size_t)(j0 + cr0) * DK + cc0);
        *(uint4*)(&Vs[buf][cr0 * FS + cc0]) = *(const uint4*)(Vh + (size_t)(j0 + cr0) * DK + cc0);
        *(uint4*)(&Ks[buf][cr1 * FS + cc1]) = *(const uint4*)(Kh + (size_t)(j0 + cr1) * DK + cc1);
        *(uint4*)(&Vs[buf][cr1 * FS + cc1]) = *(const uint4*)(Vh + (size_t)(j0 + cr1) * DK + cc1);
        __syncthreads();
#endif
        const __bf16* Kt = Ks[buf];
        const __bf16* Vt = Vs[buf];

        // S = Q @ K^T : 4 N-tiles x 2 d-chunks
        f32x8 sacc[4] = {};
#pragma unroll
        for (int t = 0; t < 4; t++) {
            const int key = t * 16 + lm;
#pragma unroll
            for (int dc = 0; dc < 2; dc++) {
                bf16x16 b;
#pragma unroll
                for (int e = 0; e < 16; e++) b[e] = Kt[key * FS + dc * 32 + kmap(e, kh)];
                sacc[t] = wmma_bf16(qa[dc], b, sacc[t]);
            }
        }
#pragma unroll
        for (int t = 0; t < 4; t++)
#pragma unroll
            for (int r = 0; r < 8; r++) sacc[t][r] *= scale;

        // online softmax; row M = r + 8*kh lives on the 16 lanes of this half
#pragma unroll
        for (int r = 0; r < 8; r++) {
            float vmax = fmaxf(fmaxf(sacc[0][r], sacc[1][r]), fmaxf(sacc[2][r], sacc[3][r]));
#pragma unroll
            for (int off = 8; off >= 1; off >>= 1)
                vmax = fmaxf(vmax, __shfl_xor(vmax, off, 16));
            const float mnew = fmaxf(mrow[r], vmax);
            const float corr = __expf(mrow[r] - mnew);
            float rsum = 0.0f;
#pragma unroll
            for (int t = 0; t < 4; t++) {
                const float p = __expf(sacc[t][r] - mnew);
                sacc[t][r] = p;
                rsum += p;
            }
#pragma unroll
            for (int off = 8; off >= 1; off >>= 1)
                rsum += __shfl_xor(rsum, off, 16);
            lrow[r] = lrow[r] * corr + rsum;
            mrow[r] = mnew;
#pragma unroll
            for (int t = 0; t < 4; t++) cacc[t][r] *= corr;
            const int prow = r + 8 * kh;
#pragma unroll
            for (int t = 0; t < 4; t++)
                Pw[prow * FS + t * 16 + lm] = (__bf16)sacc[t][r];
        }

        // context += P @ V  (same-wave LDS producer/consumer; DS ops in order)
#pragma unroll
        for (int t = 0; t < 4; t++) {
            const int d = t * 16 + lm;
#pragma unroll
            for (int kc = 0; kc < 2; kc++) {
                bf16x16 pa, vb;
#pragma unroll
                for (int e = 0; e < 16; e++) pa[e] = Pw[lm * FS + kc * 32 + kmap(e, kh)];
#pragma unroll
                for (int e = 0; e < 16; e++) vb[e] = Vt[(kc * 32 + kmap(e, kh)) * FS + d];
                cacc[t] = wmma_bf16(pa, vb, cacc[t]);
            }
        }
    }

    const int bidx = bh / NHEAD;
    const int h    = bh % NHEAD;
#pragma unroll
    for (int t = 0; t < 4; t++) {
        const int d = t * 16 + lm;
#pragma unroll
        for (int r = 0; r < 8; r++) {
            const int row = r + 8 * kh;
            const int s   = q0 + wave * 16 + row;
            const float v = cacc[t][r] / lrow[r];
            Ctx[((size_t)(bidx * SEQ + s)) * D_MODEL + h * DK + d] = (__bf16)v;
        }
    }
}

// ---------------------------------------------------------------------------
// Kernel 3: output projection. out = ctx @ Wo.T + bo, fp32 result. grid=(64,12)
// ---------------------------------------------------------------------------
__global__ __launch_bounds__(256)
void out_proj_kernel(const __bf16* __restrict__ Ctx, const float* __restrict__ Wo,
                     const float* __restrict__ bo, float* __restrict__ Out)
{
    const int m0 = blockIdx.x * 128;
    const int n0 = blockIdx.y * 64;

    __shared__ __bf16 Xs[128 * GS];
    __shared__ __bf16 Ws[64 * GS];

    const int tid  = threadIdx.x;
    const int wave = tid >> 5;
    const int lane = tid & 31;
    const int lm   = lane & 15;
    const int kh   = lane >> 4;

    f32x8 acc[4] = {};

    for (int k0 = 0; k0 < D_MODEL; k0 += 32) {
        __syncthreads();
#pragma unroll
        for (int i = 0; i < 2; i++) {           // ctx tile 128x32 bf16
            int lin = tid + i * 256;
            int r = lin >> 2, c = (lin & 3) * 8;
            *(uint4*)(&Xs[r * GS + c]) = *(const uint4*)(Ctx + (size_t)(m0 + r) * D_MODEL + k0 + c);
        }
#pragma unroll
        for (int i = 0; i < 2; i++) {           // Wo tile 64x32 f32 -> bf16
            int lin = tid + i * 256;
            int r = lin >> 3, c = (lin & 7) * 4;
            const float4 v = *(const float4*)(Wo + (size_t)(n0 + r) * D_MODEL + k0 + c);
            uint2 p; p.x = pack2bf(v.x, v.y); p.y = pack2bf(v.z, v.w);
            *(uint2*)(&Ws[r * GS + c]) = p;
        }
        __syncthreads();

        bf16x16 a;
        const int arow = wave * 16 + lm;
#pragma unroll
        for (int e = 0; e < 16; e++) a[e] = Xs[arow * GS + kmap(e, kh)];
#pragma unroll
        for (int t = 0; t < 4; t++) {
            bf16x16 b;
            const int bcol = t * 16 + lm;
#pragma unroll
            for (int e = 0; e < 16; e++) b[e] = Ws[bcol * GS + kmap(e, kh)];
            acc[t] = wmma_bf16(a, b, acc[t]);
        }
    }

#pragma unroll
    for (int t = 0; t < 4; t++) {
        const int n   = n0 + t * 16 + lm;
        const float b = bo[n];
#pragma unroll
        for (int r = 0; r < 8; r++) {
            const int M = m0 + wave * 16 + r + 8 * kh;
            Out[(size_t)M * D_MODEL + n] = acc[t][r] + b;
        }
    }
}

// ---------------------------------------------------------------------------
extern "C" void kernel_launch(void* const* d_in, const int* in_sizes, int n_in,
                              void* d_out, int out_size, void* d_ws, size_t ws_size,
                              hipStream_t stream)
{
    (void)in_sizes; (void)n_in; (void)out_size;

    const float* x  = (const float*)d_in[0];
    const float* Wq = (const float*)d_in[1];
    const float* bq = (const float*)d_in[2];
    const float* Wk = (const float*)d_in[3];
    const float* bk = (const float*)d_in[4];
    const float* Wv = (const float*)d_in[5];
    const float* bv = (const float*)d_in[6];
    const float* Wo = (const float*)d_in[7];
    const float* bo = (const float*)d_in[8];
    float* out = (float*)d_out;

    const size_t E = (size_t)MTOT * D_MODEL;      // 6,291,456 elements
    if (ws_size < 4 * E * sizeof(__bf16)) return; // ~48 MB scratch required
    __bf16* ws = (__bf16*)d_ws;
    __bf16* Qb = ws;
    __bf16* Kb = ws + E;
    __bf16* Vb = ws + 2 * E;
    __bf16* Cb = ws + 3 * E;

    dim3 g1(MTOT / 128, NHEAD, 3);
    qkv_proj_kernel<<<g1, 256, 0, stream>>>(x, Wq, bq, Wk, bk, Wv, bv, Qb, Kb, Vb);

    dim3 g2(SEQ / 128, BATCH * NHEAD);
    flash_attn_kernel<<<g2, 256, 0, stream>>>(Qb, Kb, Vb, Cb);

    dim3 g3(MTOT / 128, D_MODEL / 64);
    out_proj_kernel<<<g3, 256, 0, stream>>>(Cb, Wo, bo, out);
}
